// DecoderFirstLayer_19189913878732
// MI455X (gfx1250) — matrix operand order repaired and checked by reading, not compiled, not run on hardware
//
#include <hip/hip_runtime.h>
#include <math.h>

// ---------------------------------------------------------------------------
// DecoderFirstLayer on MI455X (gfx1250): bf16 WMMA GEMMs with LDS-staged
// B-tiles (async global->LDS copies), fused GRU gates.
// H = 1024, B = 256, T = 64, K = 1024 everywhere.
// ---------------------------------------------------------------------------

typedef __attribute__((ext_vector_type(16))) __bf16 v16bf;
typedef __attribute__((ext_vector_type(8)))  __bf16 v8bf;
typedef __attribute__((ext_vector_type(8)))  float  v8f;

#define HSZ 1024
#define BSZ 256
#define TSTEPS 64
#define KDIM 1024

// Use gfx1250 async global->LDS copies (ASYNCcnt path) for B-tile staging.
#define ASYNC_LDS 1

// float -> bf16, round-to-nearest-even
__device__ __forceinline__ __bf16 f2bf(float f) {
  unsigned u = __builtin_bit_cast(unsigned, f);
  unsigned r = u + 0x7FFFu + ((u >> 16) & 1u);
  unsigned short h = (unsigned short)(r >> 16);
  return __builtin_bit_cast(__bf16, h);
}

__device__ __forceinline__ float sigmoidf_(float x) {
  return 1.0f / (1.0f + __expf(-x));
}

// A-fragment (16x32 bf16, row-major source [M,K]): lane l holds row l&15,
// elements 0..15 = K {0..15}+16*(l>=16) (contiguous).
__device__ __forceinline__ v16bf frag_a(const __bf16* p) {
  v8bf lo = *(const v8bf*)p;
  v8bf hi = *(const v8bf*)(p + 8);
  return __builtin_shufflevector(lo, hi, 0,1,2,3,4,5,6,7,8,9,10,11,12,13,14,15);
}

// B-fragment from 16 contiguous bf16 (col l&15, K-base 16*(l>=16) layout).
__device__ __forceinline__ v16bf frag_b16(const __bf16* p) {
  v8bf lo = *(const v8bf*)p;
  v8bf hi = *(const v8bf*)(p + 8);
  return __builtin_shufflevector(lo, hi, 0,1,2,3,4,5,6,7,8,9,10,11,12,13,14,15);
}

#define WMMA_BF16(A_, B_, C_) \
  __builtin_amdgcn_wmma_f32_16x16x32_bf16(false, (A_), false, (B_), (short)0, (C_), false, false)

// Copy 16 bytes global -> LDS. Async variant uses the gfx1250 ASYNCcnt path:
// the LDS destination VGPR carries the LDS byte offset (low 32 bits of the
// generic shared-pointer), the global address rides in a VGPR pair.
__device__ __forceinline__ void stage16(const __bf16* __restrict__ g, __bf16* l) {
#if ASYNC_LDS
  unsigned lds_off = (unsigned)(uintptr_t)l;
  asm volatile("global_load_async_to_lds_b128 %0, %1, off"
               :: "v"(lds_off), "v"(g)
               : "memory");
#else
  *(v8bf*)l = *(const v8bf*)g;  // global_load_b128 + ds_store_b128
#endif
}

__device__ __forceinline__ void stage_wait_barrier() {
#if ASYNC_LDS
  asm volatile("s_wait_asynccnt 0" ::: "memory");
#endif
  __syncthreads();
}

// ---------------------------------------------------------------------------
// One-time conversion kernels
// ---------------------------------------------------------------------------

// src: f32 [K,N] row-major  ->  dst: bf16 [N,K] (N-major, WMMA-B friendly)
__global__ __launch_bounds__(256)
void transpose_to_bf16(const float* __restrict__ src, __bf16* __restrict__ dst,
                       int K, int N) {
  __shared__ float tile[32][33];
  int n0 = blockIdx.x * 32;
  int k0 = blockIdx.y * 32;
  int tx = threadIdx.x & 31;
  int ty = threadIdx.x >> 5;  // 0..7
#pragma unroll
  for (int i = 0; i < 4; ++i) {
    int k = ty + i * 8;
    tile[k][tx] = src[(size_t)(k0 + k) * N + (n0 + tx)];
  }
  __syncthreads();
#pragma unroll
  for (int i = 0; i < 4; ++i) {
    int n = ty + i * 8;
    dst[(size_t)(n0 + n) * K + (k0 + tx)] = f2bf(tile[tx][n]);
  }
}

__global__ __launch_bounds__(256)
void f32_to_bf16(const float* __restrict__ src, __bf16* __restrict__ dst, int n) {
  int i = blockIdx.x * 256 + threadIdx.x;
  if (i < n) dst[i] = f2bf(src[i]);
}

// ---------------------------------------------------------------------------
// GEMM + bias + act: C[256,1024] = act(A[256,1024] @ B + bias), act 0=relu 1=tanh
// A: bf16 row-major [256,K].  Bt: bf16 N-major [1024,K].
// Grid 64 x 256: block = (n-group of 32 cols) x (m-half of 8 M-tiles).
// Two 16x32-K B-tiles are staged in LDS per k-step and shared by all 8 waves.
// ---------------------------------------------------------------------------
__global__ __launch_bounds__(256)
void gemm_bias_act(const __bf16* __restrict__ A, const __bf16* __restrict__ Bt,
                   const float* __restrict__ bias, int act,
                   __bf16* __restrict__ out_bf,
                   float* __restrict__ out_f32, int f32_row_stride, int f32_col_off) {
  __shared__ __bf16 sB[2][2][512];  // [buf][n-tile][32 chunks x 16 bf16]

  int tid  = threadIdx.x;
  int lane = tid & 31;
  int wave = tid >> 5;              // 0..7
  int np   = blockIdx.x >> 1;       // 0..31
  int mh   = blockIdx.x & 1;        // 0..1
  int n0   = np * 32;
  int m0   = (mh * 8 + wave) * 16;
  int r15  = lane & 15;
  int hi8  = (lane >> 4) << 3;      // D row base 0/8 (A K-base 2*hi8 = 0/16)

  // Staging map: threads 0..63 each own one 32B chunk (tile st, chunk sc).
  bool do_stage = tid < 64;
  int st = tid >> 5;                // 0..1
  int sc = tid & 31;                // 0..31
  const __bf16* gsrc = Bt + (size_t)(n0 + st * 16 + (sc & 15)) * KDIM + ((sc >> 4) << 4);
  __bf16* ldst0 = &sB[0][st][sc * 16];
  __bf16* ldst1 = &sB[1][st][sc * 16];

  const __bf16* ap = A + (size_t)(m0 + r15) * KDIM + 2 * hi8;

  v8f c0 = {}; v8f c1 = {};

  if (do_stage) { stage16(gsrc, ldst0); stage16(gsrc + 8, ldst0 + 8); }

  for (int kk = 0; kk < KDIM / 32; ++kk) {
    int k0 = kk * 32;
    stage_wait_barrier();           // buf[kk&1] ready for everyone
    if (kk + 1 < KDIM / 32 && do_stage) {
      const __bf16* s = gsrc + (kk + 1) * 32;
      __bf16* d = ((kk + 1) & 1) ? ldst1 : ldst0;
      stage16(s, d); stage16(s + 8, d + 8);
    }
    v16bf a  = frag_a(ap + k0);
    const __bf16* lb = &sB[kk & 1][0][lane * 16];
    v16bf b0 = frag_b16(lb);
    v16bf b1 = frag_b16(lb + 512);
    c0 = WMMA_BF16(a, b0, c0);
    c1 = WMMA_BF16(a, b1, c1);
  }

  int col0 = n0 + r15;
  int col1 = col0 + 16;
  int rowb = m0 + hi8;              // D: lanes>=16 hold rows m0+8..m0+15
  float bias0 = bias[col0], bias1 = bias[col1];
#pragma unroll
  for (int v = 0; v < 8; ++v) {
    int row = rowb + v;
    float f0 = c0[v] + bias0;
    float f1 = c1[v] + bias1;
    if (act == 0) { f0 = fmaxf(f0, 0.0f); f1 = fmaxf(f1, 0.0f); }
    else          { f0 = tanhf(f0);       f1 = tanhf(f1);       }
    out_bf[(size_t)row * HSZ + col0] = f2bf(f0);
    out_bf[(size_t)row * HSZ + col1] = f2bf(f1);
    if (out_f32) {
      out_f32[(size_t)row * f32_row_stride + f32_col_off + col0] = f0;
      out_f32[(size_t)row * f32_row_stride + f32_col_off + col1] = f1;
    }
  }
}

// ---------------------------------------------------------------------------
// Fused GRU step (Keras reset_after=True):
//   z = sig(O@Wk_z + H@Wr_z + b) ; r = sig(O@Wk_r + H@Wr_r + b)
//   hc = tanh(O@Wk_h + b + r*(H@Wr_h + b)) ; h' = z*h + (1-z)*hc
// Grid 64 x 256: block = one 16-col tile; 8 waves x 2 M-tiles = 256 rows.
// Six 16x32-K B-tiles (Wk_z/r/h, Wr_z/r/h) staged in LDS per k-step and
// shared by all 8 waves. z/r accumulators take both GEMMs.
// ---------------------------------------------------------------------------
__global__ __launch_bounds__(256)
void gru_fused(const __bf16* __restrict__ O, const __bf16* __restrict__ Hbf,
               const float* __restrict__ Hold,
               const __bf16* __restrict__ Wk, const __bf16* __restrict__ Wr,
               const float* __restrict__ b0, const float* __restrict__ b1,
               float* __restrict__ Hnew, __bf16* __restrict__ Hbf_new) {
  __shared__ __bf16 sB[2][6][512];  // [buf][tile: kz,kr,kh,rz,rr,rh][chunks]

  int tid  = threadIdx.x;
  int lane = tid & 31;
  int wave = tid >> 5;              // 0..7 -> m-pair
  int n0   = blockIdx.x * 16;      // 0..63 n-tiles
  int m0   = wave * 32;
  int r15  = lane & 15;
  int hi8  = (lane >> 4) << 3;

  // Staging map: threads 0..191 each own one 32B chunk (tile st, chunk sc).
  bool do_stage = tid < 192;
  int st = tid >> 5;                // 0..5
  int sc = tid & 31;
  const __bf16* sbase = (st < 3) ? Wk : Wr;
  int gate = st % 3;                // z, r, h
  const __bf16* gsrc =
      sbase + (size_t)(gate * HSZ + n0 + (sc & 15)) * KDIM + ((sc >> 4) << 4);
  __bf16* ldst0 = &sB[0][st][sc * 16];
  __bf16* ldst1 = &sB[1][st][sc * 16];

  const __bf16* aoA = O   + (size_t)(m0 + r15) * KDIM + 2 * hi8;
  const __bf16* ahA = Hbf + (size_t)(m0 + r15) * KDIM + 2 * hi8;
  const __bf16* aoB = aoA + (size_t)16 * KDIM;
  const __bf16* ahB = ahA + (size_t)16 * KDIM;

  v8f za = {}, ra = {}, xha = {}, hha = {};
  v8f zb = {}, rb = {}, xhb = {}, hhb = {};

  if (do_stage) { stage16(gsrc, ldst0); stage16(gsrc + 8, ldst0 + 8); }

  for (int kk = 0; kk < KDIM / 32; ++kk) {
    int k0 = kk * 32;
    stage_wait_barrier();           // buf[kk&1] ready
    if (kk + 1 < KDIM / 32 && do_stage) {
      const __bf16* s = gsrc + (kk + 1) * 32;
      __bf16* d = ((kk + 1) & 1) ? ldst1 : ldst0;
      stage16(s, d); stage16(s + 8, d + 8);
    }

    v16bf fao = frag_a(aoA + k0);
    v16bf fah = frag_a(ahA + k0);
    v16bf fbo = frag_a(aoB + k0);
    v16bf fbh = frag_a(ahB + k0);

    const __bf16* lb = &sB[kk & 1][0][lane * 16];
    v16bf fkz = frag_b16(lb);
    v16bf fkr = frag_b16(lb + 512);
    v16bf fkh = frag_b16(lb + 1024);
    v16bf frz = frag_b16(lb + 1536);
    v16bf frr = frag_b16(lb + 2048);
    v16bf frh = frag_b16(lb + 2560);

    za  = WMMA_BF16(fao, fkz, za);  za  = WMMA_BF16(fah, frz, za);
    ra  = WMMA_BF16(fao, fkr, ra);  ra  = WMMA_BF16(fah, frr, ra);
    xha = WMMA_BF16(fao, fkh, xha); hha = WMMA_BF16(fah, frh, hha);

    zb  = WMMA_BF16(fbo, fkz, zb);  zb  = WMMA_BF16(fbh, frz, zb);
    rb  = WMMA_BF16(fbo, fkr, rb);  rb  = WMMA_BF16(fbh, frr, rb);
    xhb = WMMA_BF16(fbo, fkh, xhb); hhb = WMMA_BF16(fbh, frh, hhb);
  }

  int col = n0 + r15;
  float bz  = b0[col] + b1[col];
  float brv = b0[HSZ + col] + b1[HSZ + col];
  float bxh = b0[2 * HSZ + col];
  float bhh = b1[2 * HSZ + col];
  int rowb = m0 + hi8;

#pragma unroll
  for (int v = 0; v < 8; ++v) {
    int rA = rowb + v;
    int rB = rA + 16;
    {
      float z  = sigmoidf_(za[v] + bz);
      float r  = sigmoidf_(ra[v] + brv);
      float hc = tanhf(xha[v] + bxh + r * (hha[v] + bhh));
      float ho = Hold[(size_t)rA * HSZ + col];
      float hn = z * ho + (1.0f - z) * hc;
      Hnew[(size_t)rA * HSZ + col] = hn;
      Hbf_new[(size_t)rA * HSZ + col] = f2bf(hn);
    }
    {
      float z  = sigmoidf_(zb[v] + bz);
      float r  = sigmoidf_(rb[v] + brv);
      float hc = tanhf(xhb[v] + bxh + r * (hhb[v] + bhh));
      float ho = Hold[(size_t)rB * HSZ + col];
      float hn = z * ho + (1.0f - z) * hc;
      Hnew[(size_t)rB * HSZ + col] = hn;
      Hbf_new[(size_t)rB * HSZ + col] = f2bf(hn);
    }
  }
}

// ---------------------------------------------------------------------------
// Host launcher
// ---------------------------------------------------------------------------
extern "C" void kernel_launch(void* const* d_in, const int* in_sizes, int n_in,
                              void* d_out, int out_size, void* d_ws, size_t ws_size,
                              hipStream_t stream) {
  const float* x     = (const float*)d_in[0];
  const float* W_fs  = (const float*)d_in[1];
  const float* b_fs  = (const float*)d_in[2];
  const float* W_out = (const float*)d_in[3];
  const float* b_out = (const float*)d_in[4];
  const float* W_k   = (const float*)d_in[5];
  const float* W_r   = (const float*)d_in[6];
  const float* b_gru = (const float*)d_in[7];
  float* out = (float*)d_out;

  char* ws = (char*)d_ws;
  size_t off = 0;
  auto alloc = [&](size_t bytes) -> void* {
    void* p = ws + off;
    off += (bytes + 255) & ~(size_t)255;
    return p;
  };
  __bf16* Wt_fs  = (__bf16*)alloc((size_t)HSZ * KDIM * 2);
  __bf16* Wt_out = (__bf16*)alloc((size_t)HSZ * KDIM * 2);
  __bf16* Wt_k   = (__bf16*)alloc((size_t)3 * HSZ * KDIM * 2);
  __bf16* Wt_r   = (__bf16*)alloc((size_t)3 * HSZ * KDIM * 2);
  __bf16* x_bf   = (__bf16*)alloc((size_t)BSZ * KDIM * 2);
  float*  h_f[2];
  __bf16* h_bf[2];
  h_f[0]  = (float*)alloc((size_t)BSZ * HSZ * 4);
  h_f[1]  = (float*)alloc((size_t)BSZ * HSZ * 4);
  h_bf[0] = (__bf16*)alloc((size_t)BSZ * HSZ * 2);
  h_bf[1] = (__bf16*)alloc((size_t)BSZ * HSZ * 2);
  __bf16* o_bf = (__bf16*)alloc((size_t)BSZ * HSZ * 2);

  // One-time (per launch) weight conversion to bf16, N-major.
  transpose_to_bf16<<<dim3(HSZ / 32, KDIM / 32), 256, 0, stream>>>(W_fs, Wt_fs, KDIM, HSZ);
  transpose_to_bf16<<<dim3(HSZ / 32, KDIM / 32), 256, 0, stream>>>(W_out, Wt_out, KDIM, HSZ);
  transpose_to_bf16<<<dim3(3 * HSZ / 32, KDIM / 32), 256, 0, stream>>>(W_k, Wt_k, KDIM, 3 * HSZ);
  transpose_to_bf16<<<dim3(3 * HSZ / 32, KDIM / 32), 256, 0, stream>>>(W_r, Wt_r, KDIM, 3 * HSZ);
  f32_to_bf16<<<(BSZ * KDIM + 255) / 256, 256, 0, stream>>>(x, x_bf, BSZ * KDIM);

  // h0 = tanh(x @ W_fs + b_fs)
  gemm_bias_act<<<64, 256, 0, stream>>>(x_bf, Wt_fs, b_fs, /*act=tanh*/1,
                                        h_bf[0], h_f[0], HSZ, 0);

  // Recurrence. out_t = relu(h_t @ W_out + b_out); d_out slot (t-1) = out_t
  // for t = 1..64; GRU update for t = 0..63.
  int cur = 0;
  for (int t = 0; t <= TSTEPS; ++t) {
    float* f32dst = (t >= 1) ? out : nullptr;
    gemm_bias_act<<<64, 256, 0, stream>>>(h_bf[cur], Wt_out, b_out, /*act=relu*/0,
                                          o_bf, f32dst, TSTEPS * HSZ, (t - 1) * HSZ);
    if (t < TSTEPS) {
      gru_fused<<<64, 256, 0, stream>>>(o_bf, h_bf[cur], h_f[cur], Wt_k, Wt_r,
                                        b_gru, b_gru + 3 * HSZ,
                                        h_f[1 - cur], h_bf[1 - cur]);
      cur ^= 1;
    }
  }
}
